// NeighborhoodEncoder_44744969290073
// MI455X (gfx1250) — matrix operand order repaired and checked by reading, not compiled, not run on hardware
//
#include <hip/hip_runtime.h>
#include <hip/hip_bf16.h>

typedef __bf16 v16bf __attribute__((ext_vector_type(16)));
typedef float  v8f   __attribute__((ext_vector_type(8)));

#define D_IN  128
#define D_H1  256
#define D_H2  128
#define D_OUT 64
#define LN_EPS 1e-5f

__device__ __forceinline__ v8f wmma_bf16(v16bf a, v16bf b, v8f c) {
    // D = A(16x32 bf16) * B(32x16 bf16) + C(16x16 f32)
    return __builtin_amdgcn_wmma_f32_16x16x32_bf16(
        /*neg_a=*/false, a, /*neg_b=*/false, b,
        /*c_mod=*/(short)0, c, /*reuse_a=*/false, /*reuse_b=*/false);
}

// A-matrix (16x32, bf16) per-lane element -> K index:
//   lane half h = lane>>4, m = lane&15 ; element e: K = h*16 + e
// B-matrix (32x16, bf16): lane n = lane&15, h = lane>>4 ; element e: K = h*16 + e
// C/D (16x16 f32): VGPR r -> M = (lane>>4)*8 + r, N = lane&15

// ---------------------------------------------------------------- utilities
__global__ __launch_bounds__(256) void zero_f32_kernel(float* p, int n) {
    int i = blockIdx.x * 256 + threadIdx.x;
    if (i < n) p[i] = 0.0f;
}
__global__ __launch_bounds__(256) void cvt_bf16_kernel(const float* __restrict__ s,
                                                       __bf16* __restrict__ d, int n) {
    int i = blockIdx.x * 256 + threadIdx.x;
    if (i < n) d[i] = (__bf16)s[i];
}

// ---------------------------------------------------------------- edge stage
// pooled = relu(LN( scale_e * (x_bf[src] @ pool_w.T) + pool_b )) ; agg = segmax(dst)
// (scale is per-row, so it commutes with the GEMM: applied post-accumulate in f32)
__global__ __launch_bounds__(128)
void edge_kernel(const __bf16* __restrict__ xbf,    // [N][128] bf16 copy of x
                 const long long* __restrict__ src,
                 const long long* __restrict__ dst,
                 const float* __restrict__ ew,
                 const float* __restrict__ ewmc,
                 const __bf16* __restrict__ poolw,  // [128][128] bf16, row = out feat
                 const float* __restrict__ pool_b,
                 const float* __restrict__ lnp_g,
                 const float* __restrict__ lnp_b,
                 float* __restrict__ agg,           // [N][128], zero-initialized
                 int E)
{
    __shared__ __bf16 sA[64][D_IN];     // 16 KB: gathered bf16 rows (async DMA)
    __shared__ float  sO[64][D_IN];     // 32 KB: GEMM output (f32)
    __shared__ float  sRed[64][2][2];   // LN partial reductions
    __shared__ float  sScale[64];       // per-edge row scale (1 + sp*ew)

    const int tid  = threadIdx.x;
    const int lane = tid & 31;
    const int wave = tid >> 5;
    const int e0   = blockIdx.x * 64;

    const float mc = ewmc[0];
    const float sp = (mc > 20.f) ? mc : log1pf(expf(mc));

    // ---- gather: async global->LDS copy of bf16 rows (128 B per thread)
    {
        const int row = tid >> 1, half = tid & 1;
        int e = e0 + row; if (e >= E) e = E - 1;     // clamp: keep EXEC uniform
        const long long s = src[e];
        if (half == 0) sScale[row] = 1.f + sp * ew[e];
        const __bf16* gsrc = xbf + (size_t)s * D_IN + half * 64;
        unsigned lds = (unsigned)(uintptr_t)&sA[row][half * 64];
        #pragma unroll
        for (int q = 0; q < 8; ++q) {
            const void* ga = (const void*)(gsrc + q * 8);   // 16 B per op
            asm volatile("global_load_async_to_lds_b128 %0, %1, off"
                         :: "v"(lds + q * 16), "v"(ga)
                         : "memory");
        }
        asm volatile("s_wait_asynccnt 0" ::: "memory");
    }
    __syncthreads();

    // ---- WMMA: each wave computes 16 rows x 128 cols (8 n-tiles, 4 k-steps)
    {
        const int m0  = wave * 16;
        const int am  = m0 + (lane & 15);
        const int khb = (lane >> 4) << 4;   // A K sub-offset
        const int bn  = lane & 15;
        const int bkb = (lane >> 4) << 4;   // B K sub-offset
        for (int nt = 0; nt < 8; ++nt) {
            v8f acc = {};
            #pragma unroll
            for (int kt = 0; kt < 4; ++kt) {
                const int k0 = kt * 32;
                v16bf a, b;
                const __bf16* ap = &sA[am][k0 + khb];
                #pragma unroll
                for (int q = 0; q < 8; ++q) { a[q] = ap[q]; a[q + 8] = ap[8 + q]; }
                const __bf16* bp = poolw + (size_t)(nt * 16 + bn) * D_IN + k0 + bkb;
                #pragma unroll
                for (int q = 0; q < 16; ++q) b[q] = bp[q];
                acc = wmma_bf16(a, b, acc);
            }
            const int orow = m0 + ((lane >> 4) << 3);
            const int ocol = nt * 16 + (lane & 15);
            #pragma unroll
            for (int r = 0; r < 8; ++r) sO[orow + r][ocol] = acc[r];
        }
    }
    __syncthreads();

    // ---- scale + bias + LayerNorm + ReLU + scatter-max (uint trick: values >= 0)
    {
        const int row = tid >> 1, half = tid & 1;
        const int jb = half * 64;
        const float sc = sScale[row];
        float sum = 0.f, sq = 0.f;
        #pragma unroll 4
        for (int c = 0; c < 64; ++c) {
            float v = sO[row][jb + c] * sc + pool_b[jb + c];
            sum += v; sq += v * v;
        }
        sRed[row][half][0] = sum; sRed[row][half][1] = sq;
        __syncthreads();
        const float S = sRed[row][0][0] + sRed[row][1][0];
        const float Q = sRed[row][0][1] + sRed[row][1][1];
        const float mean = S * (1.f / 128.f);
        const float var  = Q * (1.f / 128.f) - mean * mean;
        const float inv  = rsqrtf(var + LN_EPS);
        const int e = e0 + row;
        if (e < E) {
            const long long d = dst[e];
            unsigned int* ar = (unsigned int*)(agg + (size_t)d * D_IN);
            for (int c = 0; c < 64; ++c) {
                const int j = jb + c;
                float v = (sO[row][j] * sc + pool_b[j] - mean) * inv * lnp_g[j] + lnp_b[j];
                v = fmaxf(v, 0.f);
                atomicMax(&ar[j], __float_as_uint(v));
            }
        }
    }
}

// ---------------------------------------------------------------- node stage
__global__ __launch_bounds__(128)
void node_kernel(const float* __restrict__ agg,
                 const float* __restrict__ ln0_g, const float* __restrict__ ln0_b,
                 const __bf16* __restrict__ w1,   const float* __restrict__ b1,
                 const float* __restrict__ ln1_g, const float* __restrict__ ln1_b,
                 const __bf16* __restrict__ w2,   const float* __restrict__ b2,
                 const float* __restrict__ ln2_g, const float* __restrict__ ln2_b,
                 const __bf16* __restrict__ muw,  const float* __restrict__ mub,
                 const __bf16* __restrict__ lvw,  const float* __restrict__ lvb,
                 float* __restrict__ out, int N)
{
    __shared__ __bf16 sHa[64][D_IN];    // 16 KB (h0, later h2)
    __shared__ __bf16 sHb[64][D_H1];    // 32 KB (h1)
    __shared__ float  sRed[64][2][2];
    __shared__ float  sStatA[64][2];    // per-row sum / sumsq (GEMM1)
    __shared__ float  sStatB[64][2];    // per-row sum / sumsq (GEMM2)

    const int tid   = threadIdx.x;
    const int lane  = tid & 31;
    const int wave  = tid >> 5;
    const int n0    = blockIdx.x * 64;
    const int row2  = tid >> 1, half2 = tid & 1;

    const int m0  = wave * 16;
    const int am  = m0 + (lane & 15);
    const int khb = (lane >> 4) << 4;
    const int bn  = lane & 15;
    const int bkb = (lane >> 4) << 4;
    const int rbase = m0 + ((lane >> 4) << 3);

    // ---- Stage A: h0 = LN0(agg) -> bf16 (agg >= 0 already handles clamp)
    {
        const int node = n0 + row2;
        const float* ar = agg + (size_t)node * D_IN + half2 * 64;
        float sum = 0.f, sq = 0.f;
        if (node < N) {
            #pragma unroll 4
            for (int c = 0; c < 64; ++c) { float v = ar[c]; sum += v; sq += v * v; }
        }
        sRed[row2][half2][0] = sum; sRed[row2][half2][1] = sq;
        if (tid < 64) { sStatA[tid][0] = 0.f; sStatA[tid][1] = 0.f;
                        sStatB[tid][0] = 0.f; sStatB[tid][1] = 0.f; }
        __syncthreads();
        const float S = sRed[row2][0][0] + sRed[row2][1][0];
        const float Q = sRed[row2][0][1] + sRed[row2][1][1];
        const float mean = S * (1.f / 128.f);
        const float inv  = rsqrtf(Q * (1.f / 128.f) - mean * mean + LN_EPS);
        for (int c = 0; c < 64; ++c) {
            const int j = half2 * 64 + c;
            float v = (node < N) ? ar[c] : 0.f;
            sHa[row2][j] = (__bf16)((v - mean) * inv * ln0_g[j] + ln0_b[j]);
        }
    }
    __syncthreads();

    // ---- Stage B: h1_raw = h0 @ w1.T + b1  (64x128 @ 128x256), stats in f32
    {
        float sums[8], sqs[8];
        #pragma unroll
        for (int r = 0; r < 8; ++r) { sums[r] = 0.f; sqs[r] = 0.f; }
        for (int nt = 0; nt < 16; ++nt) {
            v8f acc = {};
            #pragma unroll
            for (int kt = 0; kt < 4; ++kt) {
                const int k0 = kt * 32;
                v16bf a, b;
                const __bf16* ap = &sHa[am][k0 + khb];
                #pragma unroll
                for (int q = 0; q < 8; ++q) { a[q] = ap[q]; a[q + 8] = ap[8 + q]; }
                const __bf16* bp = w1 + (size_t)(nt * 16 + bn) * D_IN + k0 + bkb;
                #pragma unroll
                for (int q = 0; q < 16; ++q) b[q] = bp[q];
                acc = wmma_bf16(a, b, acc);
            }
            const int ocol = nt * 16 + (lane & 15);
            #pragma unroll
            for (int r = 0; r < 8; ++r) {
                float v = acc[r] + b1[ocol];
                sHb[rbase + r][ocol] = (__bf16)v;
                sums[r] += v; sqs[r] += v * v;
            }
        }
        #pragma unroll
        for (int r = 0; r < 8; ++r) {
            atomicAdd(&sStatA[rbase + r][0], sums[r]);
            atomicAdd(&sStatA[rbase + r][1], sqs[r]);
        }
    }
    __syncthreads();

    // ---- Stage C: h1 = relu(LN1(h1_raw)) in place (bf16)
    {
        const float mean = sStatA[row2][0] * (1.f / 256.f);
        const float inv  = rsqrtf(sStatA[row2][1] * (1.f / 256.f) - mean * mean + LN_EPS);
        const int jb = half2 * 128;
        for (int c = 0; c < 128; ++c) {
            const int j = jb + c;
            float v = (float)sHb[row2][j];
            v = fmaxf((v - mean) * inv * ln1_g[j] + ln1_b[j], 0.f);
            sHb[row2][j] = (__bf16)v;
        }
    }
    __syncthreads();

    // ---- Stage D: h2_raw = h1 @ w2.T + b2  (64x256 @ 256x128) -> sHa
    {
        float sums[8], sqs[8];
        #pragma unroll
        for (int r = 0; r < 8; ++r) { sums[r] = 0.f; sqs[r] = 0.f; }
        for (int nt = 0; nt < 8; ++nt) {
            v8f acc = {};
            #pragma unroll
            for (int kt = 0; kt < 8; ++kt) {
                const int k0 = kt * 32;
                v16bf a, b;
                const __bf16* ap = &sHb[am][k0 + khb];
                #pragma unroll
                for (int q = 0; q < 8; ++q) { a[q] = ap[q]; a[q + 8] = ap[8 + q]; }
                const __bf16* bp = w2 + (size_t)(nt * 16 + bn) * D_H1 + k0 + bkb;
                #pragma unroll
                for (int q = 0; q < 16; ++q) b[q] = bp[q];
                acc = wmma_bf16(a, b, acc);
            }
            const int ocol = nt * 16 + (lane & 15);
            #pragma unroll
            for (int r = 0; r < 8; ++r) {
                float v = acc[r] + b2[ocol];
                sHa[rbase + r][ocol] = (__bf16)v;
                sums[r] += v; sqs[r] += v * v;
            }
        }
        #pragma unroll
        for (int r = 0; r < 8; ++r) {
            atomicAdd(&sStatB[rbase + r][0], sums[r]);
            atomicAdd(&sStatB[rbase + r][1], sqs[r]);
        }
    }
    __syncthreads();

    // ---- Stage E: h2 = relu(LN2(h2_raw)) in place (bf16)
    {
        const float mean = sStatB[row2][0] * (1.f / 128.f);
        const float inv  = rsqrtf(sStatB[row2][1] * (1.f / 128.f) - mean * mean + LN_EPS);
        const int jb = half2 * 64;
        for (int c = 0; c < 64; ++c) {
            const int j = jb + c;
            float v = (float)sHa[row2][j];
            v = fmaxf((v - mean) * inv * ln2_g[j] + ln2_b[j], 0.f);
            sHa[row2][j] = (__bf16)v;
        }
    }
    __syncthreads();

    // ---- Stage F: heads. mu = h2@mu_w.T + mu_b ; std = exp(0.5*(h2@lv_w.T + lv_b))
    {
        float* mu_out = out;
        float* sd_out = out + (size_t)N * D_OUT;
        for (int nt = 0; nt < 4; ++nt) {
            v8f accm = {}, accv = {};
            #pragma unroll
            for (int kt = 0; kt < 4; ++kt) {
                const int k0 = kt * 32;
                v16bf a, bm, bv;
                const __bf16* ap = &sHa[am][k0 + khb];
                #pragma unroll
                for (int q = 0; q < 8; ++q) { a[q] = ap[q]; a[q + 8] = ap[8 + q]; }
                const __bf16* bpm = muw + (size_t)(nt * 16 + bn) * D_H2 + k0 + bkb;
                const __bf16* bpv = lvw + (size_t)(nt * 16 + bn) * D_H2 + k0 + bkb;
                #pragma unroll
                for (int q = 0; q < 16; ++q) { bm[q] = bpm[q]; bv[q] = bpv[q]; }
                accm = wmma_bf16(a, bm, accm);
                accv = wmma_bf16(a, bv, accv);
            }
            const int col = nt * 16 + (lane & 15);
            #pragma unroll
            for (int r = 0; r < 8; ++r) {
                const int node = n0 + rbase + r;
                if (node < N) {
                    mu_out[(size_t)node * D_OUT + col] = accm[r] + mub[col];
                    sd_out[(size_t)node * D_OUT + col] = expf(0.5f * (accv[r] + lvb[col]));
                }
            }
        }
    }
}

// ---------------------------------------------------------------- launch
extern "C" void kernel_launch(void* const* d_in, const int* in_sizes, int n_in,
                              void* d_out, int out_size, void* d_ws, size_t ws_size,
                              hipStream_t stream) {
    (void)n_in; (void)out_size; (void)ws_size;
    const float*     x      = (const float*)d_in[0];
    const long long* ei     = (const long long*)d_in[1];   // int64 [2][E]
    const float*     ew     = (const float*)d_in[2];
    const float*     ewmc   = (const float*)d_in[3];
    const float*     pool_w = (const float*)d_in[4];
    const float*     pool_b = (const float*)d_in[5];
    const float*     lnp_g  = (const float*)d_in[6];
    const float*     lnp_b  = (const float*)d_in[7];
    const float*     ln0_g  = (const float*)d_in[8];
    const float*     ln0_b  = (const float*)d_in[9];
    const float*     w1     = (const float*)d_in[10];
    const float*     b1     = (const float*)d_in[11];
    const float*     ln1_g  = (const float*)d_in[12];
    const float*     ln1_b  = (const float*)d_in[13];
    const float*     w2     = (const float*)d_in[14];
    const float*     b2     = (const float*)d_in[15];
    const float*     ln2_g  = (const float*)d_in[16];
    const float*     ln2_b  = (const float*)d_in[17];
    const float*     mu_w   = (const float*)d_in[18];
    const float*     mu_b   = (const float*)d_in[19];
    const float*     lv_w   = (const float*)d_in[20];
    const float*     lv_b   = (const float*)d_in[21];

    const int N = in_sizes[0] / D_IN;   // 50000 nodes
    const int E = in_sizes[2];          // 800000 edges

    // workspace layout: agg f32 [N*128] | bf16 weight copies | bf16 x copy
    char*  ws  = (char*)d_ws;
    float* agg = (float*)ws;
    size_t off = ((size_t)N * D_IN * sizeof(float) + 255) & ~(size_t)255;
    __bf16* poolw_bf = (__bf16*)(ws + off); off += (size_t)D_IN  * D_IN * 2;
    __bf16* w1_bf    = (__bf16*)(ws + off); off += (size_t)D_H1  * D_IN * 2;
    __bf16* w2_bf    = (__bf16*)(ws + off); off += (size_t)D_H2  * D_H1 * 2;
    __bf16* muw_bf   = (__bf16*)(ws + off); off += (size_t)D_OUT * D_H2 * 2;
    __bf16* lvw_bf   = (__bf16*)(ws + off); off += (size_t)D_OUT * D_H2 * 2;
    off = (off + 255) & ~(size_t)255;
    __bf16* x_bf     = (__bf16*)(ws + off); off += (size_t)N * D_IN * 2;

    const int aggN = N * D_IN;
    zero_f32_kernel<<<(aggN + 255) / 256, 256, 0, stream>>>(agg, aggN);
    cvt_bf16_kernel<<<(D_IN  * D_IN + 255) / 256, 256, 0, stream>>>(pool_w, poolw_bf, D_IN  * D_IN);
    cvt_bf16_kernel<<<(D_H1  * D_IN + 255) / 256, 256, 0, stream>>>(w1,     w1_bf,    D_H1  * D_IN);
    cvt_bf16_kernel<<<(D_H2  * D_H1 + 255) / 256, 256, 0, stream>>>(w2,     w2_bf,    D_H2  * D_H1);
    cvt_bf16_kernel<<<(D_OUT * D_H2 + 255) / 256, 256, 0, stream>>>(mu_w,   muw_bf,   D_OUT * D_H2);
    cvt_bf16_kernel<<<(D_OUT * D_H2 + 255) / 256, 256, 0, stream>>>(lv_w,   lvw_bf,   D_OUT * D_H2);
    cvt_bf16_kernel<<<(aggN + 255) / 256, 256, 0, stream>>>(x, x_bf, aggN);

    edge_kernel<<<(E + 63) / 64, 128, 0, stream>>>(
        x_bf, ei, ei + E, ew, ewmc, poolw_bf, pool_b, lnp_g, lnp_b, agg, E);

    node_kernel<<<(N + 63) / 64, 128, 0, stream>>>(
        agg, ln0_g, ln0_b, w1_bf, b1, ln1_g, ln1_b,
        w2_bf, b2, ln2_g, ln2_b, muw_bf, mu_b, lvw_bf, lv_b,
        (float*)d_out, N);
}
